// Flux_46213848105829
// MI455X (gfx1250) — compile-verified, matrix operand order checked by reading, not run on hardware
//
#include <hip/hip_runtime.h>
#include <hip/hip_bf16.h>
#include <math.h>

// ---------------------------------------------------------------------------
// Types for CDNA5 WMMA (wave32): v_wmma_f32_16x16x32_bf16
// ---------------------------------------------------------------------------
typedef __attribute__((ext_vector_type(16))) __bf16 v16bf;
typedef __attribute__((ext_vector_type(8)))  __bf16 v8bf;
typedef __attribute__((ext_vector_type(4)))  __bf16 v4bf;
typedef __attribute__((ext_vector_type(8)))  float  v8f;
typedef __attribute__((ext_vector_type(4)))  unsigned int v4u;
typedef __attribute__((ext_vector_type(8)))  int v8i;
typedef __attribute__((ext_vector_type(4)))  int v4i;

#define LDS_STRIDE 40   // bf16 P-tile stride in flash kernel (80B rows)
#define FSTRIDE    36   // fp32 GEMM tile stride: 32 + 4 pad DWORDs (TDM pad) -> conflict-free

#if defined(__gfx1250__) && __has_builtin(__builtin_amdgcn_tensor_load_to_lds) && \
    __has_builtin(__builtin_amdgcn_s_wait_tensorcnt)
#define HAVE_TDM 1
#else
#define HAVE_TDM 0
#endif

// Build a 16x32 bf16 A/B fragment for wmma_f32_16x16x32_bf16 from bf16 memory.
// Per ISA: lane holds K chunks [half*8, half*8+8) and [16+half*8, 16+half*8+8).
// Caller passes pointer to (row, half*8); second chunk is +16 elements.
__device__ __forceinline__ v16bf frag_from(const __bf16* p) {
  v8bf lo = *(const v8bf*)(p);
  v8bf hi = *(const v8bf*)(p + 16);
  v16bf f;
#pragma unroll
  for (int i = 0; i < 8; ++i) { f[i] = lo[i]; f[8 + i] = hi[i]; }
  return f;
}

// Same fragment built from an fp32 LDS tile (cvt at fragment build; TDM copies raw).
__device__ __forceinline__ v16bf frag_f32(const float* p) {
  v16bf f;
#pragma unroll
  for (int i = 0; i < 8; ++i) { f[i] = (__bf16)p[i]; f[8 + i] = (__bf16)p[16 + i]; }
  return f;
}

__device__ __forceinline__ float gelu_f(float x) {
  float x3 = x * x * x;
  return 0.5f * x * (1.f + tanhf(0.7978845608028654f * (x + 0.044715f * x3)));
}
__device__ __forceinline__ float silu_f(float x) { return x / (1.f + __expf(-x)); }

#if HAVE_TDM
// Issue one TDM 2D tile load: 32 cols x 128 rows of fp32, HBM -> LDS, with
// 4-DWORD padding per 32-DWORD row (LDS row stride = 36 floats) and hardware
// OOB zero-fill for row/col tails. Descriptor packing per CDNA5 ISA ch.8.
__device__ __forceinline__ void tdm_load_tile(unsigned lds_byte_addr, const float* gsrc,
                                              int rows_rem, int cols_rem, long long lda_elems) {
  unsigned long long ga = (unsigned long long)(size_t)gsrc;
  v4u g0;
  g0[0] = 1u;                                   // count=1 (valid user descriptor)
  g0[1] = lds_byte_addr;                        // lds_addr
  g0[2] = (unsigned)ga;                         // global_addr[31:0]
  g0[3] = (unsigned)(ga >> 32) | (2u << 30);    // global_addr[56:32] | type=2
  unsigned td0 = (unsigned)cols_rem;            // tensor_dim0 (elements)
  unsigned td1 = (unsigned)(rows_rem < 0 ? 0 : rows_rem);  // tensor_dim1 (rows)
  v8i g1;
  g1[0] = (2 << 16)        // data_size = 4B
        | (1 << 20)        // pad_enable
        | (4 << 22)        // pad_interval: every 32 DWORDs
        | (3 << 25);       // pad_amount: 4 DWORDs
  g1[1] = (int)((td0 & 0xffffu) << 16);                    // tensor_dim0 lo16 @ [63:48]
  g1[2] = (int)((td0 >> 16) | ((td1 & 0xffffu) << 16));    // dim0 hi16, dim1 lo16
  g1[3] = (int)((td1 >> 16) | (32u << 16));                // dim1 hi16, tile_dim0=32
  g1[4] = 128;                                             // tile_dim1=128, tile_dim2=0
  g1[5] = (int)(unsigned)(lda_elems & 0xffffffffll);       // tensor_dim0_stride lo32
  g1[6] = (int)((unsigned)(lda_elems >> 32) & 0xffffu);    // dim0_stride hi16
  g1[7] = 0;
  v4i gz = {0, 0, 0, 0};
#if __clang_major__ >= 23
  v8i gz8 = {0, 0, 0, 0, 0, 0, 0, 0};
  __builtin_amdgcn_tensor_load_to_lds(g0, g1, gz, gz, gz8, 0);
#else
  __builtin_amdgcn_tensor_load_to_lds(g0, g1, gz, gz, 0);
#endif
}
#endif  // HAVE_TDM

// ---------------------------------------------------------------------------
// Generic NT GEMM: C[M,N] = epi( A[M,K] @ W[N,K]^T + bias[N] )
//   ACT: 0=none 1=gelu 2=silu  (applied before gate/residual)
//   if res != null: C = res + (gate ? gate[n]*val : val)
// Tiles staged HBM->LDS by the Tensor Data Mover (double-buffered, async,
// TENSORcnt), fp32 in LDS, converted to bf16 at fragment build, WMMA bf16.
// Block: 256 thr (8 waves), tile 128x128, BK=32. Wave tile 32x64 (2x4 wmma).
// ---------------------------------------------------------------------------
template <int ACT>
__global__ __launch_bounds__(256) void gemm_nt(
    const float* __restrict__ A, int lda,
    const float* __restrict__ W,
    const float* __restrict__ bias,
    const float* res,
    const float* __restrict__ gate,
    float* C, int ldc,
    int M, int N, int K) {
#if HAVE_TDM
  __shared__ __align__(16) float Asf[2][128 * FSTRIDE];
  __shared__ __align__(16) float Wsf[2][128 * FSTRIDE];
#else
  __shared__ __align__(16) float Asf[1][128 * FSTRIDE];
  __shared__ __align__(16) float Wsf[1][128 * FSTRIDE];
#endif
  const int tid  = threadIdx.x;
  const int wave = tid >> 5, lane = tid & 31;
  const int half = lane >> 4, r = lane & 15;
  const int wm = wave & 3;   // 4 waves along M (32 rows each)
  const int wn = wave >> 2;  // 2 waves along N (64 cols each)
  const int m0 = blockIdx.y * 128, n0 = blockIdx.x * 128;

  v8f acc[2][4];
#pragma unroll
  for (int i = 0; i < 2; ++i)
#pragma unroll
    for (int j = 0; j < 4; ++j)
#pragma unroll
      for (int e = 0; e < 8; ++e) acc[i][j][e] = 0.f;

  auto compute = [&](const float* At, const float* Wt) {
    v16bf af[2], bfr[4];
#pragma unroll
    for (int i = 0; i < 2; ++i)
      af[i] = frag_f32(&At[(wm * 32 + i * 16 + r) * FSTRIDE + half * 8]);
#pragma unroll
    for (int j = 0; j < 4; ++j)
      bfr[j] = frag_f32(&Wt[(wn * 64 + j * 16 + r) * FSTRIDE + half * 8]);
#pragma unroll
    for (int i = 0; i < 2; ++i)
#pragma unroll
      for (int j = 0; j < 4; ++j)
        acc[i][j] = __builtin_amdgcn_wmma_f32_16x16x32_bf16(
            false, af[i], false, bfr[j], (short)0, acc[i][j], false, false);
  };

#if HAVE_TDM
  const int nk = K >> 5;
  if (tid < 32) {  // wave 0 drives the DMA (EXEC-independent, one op per wave)
    tdm_load_tile((unsigned)(size_t)&Asf[0][0], A + (size_t)m0 * lda, M - m0, K, lda);
    tdm_load_tile((unsigned)(size_t)&Wsf[0][0], W + (size_t)n0 * K, N - n0, K, K);
  }
  for (int t = 0; t < nk; ++t) {
    const int p = t & 1;
    if (tid < 32) __builtin_amdgcn_s_wait_tensorcnt(0);
    __syncthreads();  // tile t visible to all; all waves done with buffer p^1
    if ((t + 1 < nk) && tid < 32) {
      const int kk = (t + 1) << 5;
      tdm_load_tile((unsigned)(size_t)&Asf[p ^ 1][0], A + (size_t)m0 * lda + kk,
                    M - m0, K - kk, lda);
      tdm_load_tile((unsigned)(size_t)&Wsf[p ^ 1][0], W + (size_t)n0 * K + kk,
                    N - n0, K - kk, K);
    }
    compute(Asf[p], Wsf[p]);  // overlaps with TDM fill of buffer p^1
  }
#else
  for (int kk = 0; kk < K; kk += 32) {
    __syncthreads();
#pragma unroll
    for (int i = 0; i < 4; ++i) {
      int f   = tid + 256 * i;  // 1024 float4 slots; 8 per row
      int row = f >> 3, c4 = f & 7;
      float4 av = make_float4(0.f, 0.f, 0.f, 0.f);
      float4 wv = make_float4(0.f, 0.f, 0.f, 0.f);
      int gm = m0 + row, gn = n0 + row;
      if (gm < M) av = *(const float4*)(A + (size_t)gm * lda + kk + c4 * 4);
      if (gn < N) wv = *(const float4*)(W + (size_t)gn * K + kk + c4 * 4);
      *(float4*)&Asf[0][row * FSTRIDE + c4 * 4] = av;
      *(float4*)&Wsf[0][row * FSTRIDE + c4 * 4] = wv;
    }
    __syncthreads();
    compute(Asf[0], Wsf[0]);
  }
#endif

  // Epilogue. C/D layout: m_local = v + 8*half, n_local = lane&15.
#pragma unroll
  for (int i = 0; i < 2; ++i) {
#pragma unroll
    for (int j = 0; j < 4; ++j) {
#pragma unroll
      for (int v = 0; v < 8; ++v) {
        int mrow = m0 + wm * 32 + i * 16 + v + 8 * half;
        int ncol = n0 + wn * 64 + j * 16 + r;
        if (mrow < M && ncol < N) {
          float val = acc[i][j][v] + bias[ncol];
          if (ACT == 1) val = gelu_f(val);
          else if (ACT == 2) val = silu_f(val);
          if (res) val = res[(size_t)mrow * ldc + ncol] + (gate ? gate[ncol] * val : val);
          C[(size_t)mrow * ldc + ncol] = val;
        }
      }
    }
  }
}

// ---------------------------------------------------------------------------
// Row LayerNorm with modulation: out = (1+sc[c]) * LN(x) + sh[c]
// ---------------------------------------------------------------------------
__global__ __launch_bounds__(256) void ln_mod(
    const float* __restrict__ x, float* __restrict__ out,
    const float* __restrict__ sc, const float* __restrict__ sh, int dim) {
  __shared__ float r1[256], r2[256];
  const int row = blockIdx.x, tid = threadIdx.x;
  const float* xr = x + (size_t)row * dim;
  float s = 0.f, s2 = 0.f;
  for (int c = tid; c < dim; c += 256) { float v = xr[c]; s += v; s2 += v * v; }
  r1[tid] = s; r2[tid] = s2; __syncthreads();
  for (int o = 128; o > 0; o >>= 1) {
    if (tid < o) { r1[tid] += r1[tid + o]; r2[tid] += r2[tid + o]; }
    __syncthreads();
  }
  float mean = r1[0] / dim;
  float var  = r2[0] / dim - mean * mean;
  float rstd = rsqrtf(var + 1e-6f);
  float* orow = out + (size_t)row * dim;
  for (int c = tid; c < dim; c += 256)
    orow[c] = (1.f + sc[c]) * ((xr[c] - mean) * rstd) + sh[c];
}

// Row RMSNorm with scale: out = x * rsqrt(mean(x^2)+eps) * scale[c]
__global__ __launch_bounds__(256) void rms_row(
    const float* __restrict__ x, float* __restrict__ out,
    const float* __restrict__ scale, int dim) {
  __shared__ float r1[256];
  const int row = blockIdx.x, tid = threadIdx.x;
  const float* xr = x + (size_t)row * dim;
  float s2 = 0.f;
  for (int c = tid; c < dim; c += 256) { float v = xr[c]; s2 += v * v; }
  r1[tid] = s2; __syncthreads();
  for (int o = 128; o > 0; o >>= 1) {
    if (tid < o) r1[tid] += r1[tid + o];
    __syncthreads();
  }
  float rr = rsqrtf(r1[0] / dim + 1e-6f);
  float* orow = out + (size_t)row * dim;
  for (int c = tid; c < dim; c += 256) orow[c] = xr[c] * rr * scale[c];
}

// ---------------------------------------------------------------------------
// QKV post-process: per (token, head) RMS-norm q/k, apply RoPE, pack bf16.
//   qb,kb: [H][Ltot][128] (roped);  vt: [H][128][Ltot] (transposed for P*V B-frags)
// qkv row layout: l*ld + s*2048 + h*128 + d (s = 0:q 1:k 2:v)
// ---------------------------------------------------------------------------
__global__ __launch_bounds__(128) void qkv_post(
    const float* __restrict__ qkv, int ld, int posoff,
    const float* __restrict__ qs, const float* __restrict__ ksc,
    const float* __restrict__ pe,
    __bf16* __restrict__ qb, __bf16* __restrict__ kb, __bf16* __restrict__ vt,
    int Ltot) {
  const int l = blockIdx.x, h = blockIdx.y, d = threadIdx.x;
  const float* base = qkv + (size_t)l * ld + h * 128 + d;
  float q = base[0], k = base[2048], v = base[4096];
  __shared__ float r1[128], r2[128];
  r1[d] = q * q; r2[d] = k * k; __syncthreads();
  for (int o = 64; o > 0; o >>= 1) {
    if (d < o) { r1[d] += r1[d + o]; r2[d] += r2[d + o]; }
    __syncthreads();
  }
  float qn = q * rsqrtf(r1[0] / 128.f + 1e-6f) * qs[d];
  float kn = k * rsqrtf(r2[0] / 128.f + 1e-6f) * ksc[d];
  int pos = posoff + l;
  // pe[pos][i][2][2]: [cos, -sin; sin, cos]
  const float* pb = pe + ((size_t)pos * 64 + (d >> 1)) * 4;
  float qp = __shfl_xor(qn, 1), kp = __shfl_xor(kn, 1);
  float qe, qo, ke, ko;
  if (d & 1) { qe = qp; qo = qn; ke = kp; ko = kn; }
  else       { qe = qn; qo = qp; ke = kn; ko = kp; }
  float qr = (d & 1) ? (pb[2] * qe + pb[3] * qo) : (pb[0] * qe + pb[1] * qo);
  float kr = (d & 1) ? (pb[2] * ke + pb[3] * ko) : (pb[0] * ke + pb[1] * ko);
  size_t o = ((size_t)h * Ltot + pos) * 128 + d;
  qb[o] = (__bf16)qr;
  kb[o] = (__bf16)kr;
  vt[((size_t)h * 128 + d) * Ltot + pos] = (__bf16)v;
}

// ---------------------------------------------------------------------------
// Flash attention (bf16 WMMA for QK^T and P*V, online softmax).
// Grid: (L/64, H). Block: 128 thr = 4 waves; wave owns 16 q-rows, all 128 d.
// ---------------------------------------------------------------------------
__global__ __launch_bounds__(128) void flash_attn(
    const __bf16* __restrict__ qb, const __bf16* __restrict__ kb,
    const __bf16* __restrict__ vt, float* __restrict__ out, int L) {
  __shared__ __align__(16) __bf16 Pl[4 * 16 * LDS_STRIDE];
  const int tid = threadIdx.x;
  const int w = tid >> 5, lane = tid & 31, half = lane >> 4, r = lane & 15;
  const int h = blockIdx.y;
  const int q0 = blockIdx.x * 64;
  const float scl = 0.08838834764831845f;  // 1/sqrt(128)

  const __bf16* qrow = qb + ((size_t)h * L + q0 + w * 16 + r) * 128;
  v16bf qf[4];
#pragma unroll
  for (int ks = 0; ks < 4; ++ks) qf[ks] = frag_from(qrow + ks * 32 + half * 8);

  v8f acc[8];
  float mrun[8], lrun[8];
#pragma unroll
  for (int dt = 0; dt < 8; ++dt)
#pragma unroll
    for (int e = 0; e < 8; ++e) acc[dt][e] = 0.f;
#pragma unroll
  for (int v = 0; v < 8; ++v) { mrun[v] = -1e30f; lrun[v] = 0.f; }

  __bf16* myP = Pl + w * 16 * LDS_STRIDE;
  const __bf16* kbh = kb + (size_t)h * L * 128;
  const __bf16* vth = vt + (size_t)h * 128 * L;

  for (int c = 0; c < L; c += 32) {
    v8f s0, s1;
#pragma unroll
    for (int e = 0; e < 8; ++e) { s0[e] = 0.f; s1[e] = 0.f; }
#pragma unroll
    for (int ks = 0; ks < 4; ++ks) {
      v16bf k0 = frag_from(kbh + (size_t)(c + r) * 128 + ks * 32 + half * 8);
      v16bf k1 = frag_from(kbh + (size_t)(c + 16 + r) * 128 + ks * 32 + half * 8);
      s0 = __builtin_amdgcn_wmma_f32_16x16x32_bf16(false, qf[ks], false, k0,
                                                   (short)0, s0, false, false);
      s1 = __builtin_amdgcn_wmma_f32_16x16x32_bf16(false, qf[ks], false, k1,
                                                   (short)0, s1, false, false);
    }
    // Online softmax per row (row m = v + 8*half lives on 16 lanes of a half).
    float al[8];
#pragma unroll
    for (int v = 0; v < 8; ++v) {
      float a = s0[v] * scl, b = s1[v] * scl;
      float cm = fmaxf(a, b);
      cm = fmaxf(cm, __shfl_xor(cm, 1));
      cm = fmaxf(cm, __shfl_xor(cm, 2));
      cm = fmaxf(cm, __shfl_xor(cm, 4));
      cm = fmaxf(cm, __shfl_xor(cm, 8));
      float nm = fmaxf(mrun[v], cm);
      al[v] = __expf(mrun[v] - nm);
      mrun[v] = nm;
      float p0 = __expf(a - nm), p1 = __expf(b - nm);
      s0[v] = p0; s1[v] = p1;
      float rs = p0 + p1;
      rs += __shfl_xor(rs, 1);
      rs += __shfl_xor(rs, 2);
      rs += __shfl_xor(rs, 4);
      rs += __shfl_xor(rs, 8);
      lrun[v] = lrun[v] * al[v] + rs;
    }
#pragma unroll
    for (int dt = 0; dt < 8; ++dt)
#pragma unroll
      for (int v = 0; v < 8; ++v) acc[dt][v] *= al[v];

    // C-layout P tile -> LDS (row-major 16x32, stride 40) -> A-fragment.
#pragma unroll
    for (int v = 0; v < 8; ++v) {
      int m = v + 8 * half;
      myP[m * LDS_STRIDE + r] = (__bf16)s0[v];
      myP[m * LDS_STRIDE + 16 + r] = (__bf16)s1[v];
    }
    __syncthreads();
    v16bf pf = frag_from(myP + r * LDS_STRIDE + half * 8);
#pragma unroll
    for (int dt = 0; dt < 8; ++dt) {
      v16bf vf = frag_from(vth + (size_t)(dt * 16 + r) * L + c + half * 8);
      acc[dt] = __builtin_amdgcn_wmma_f32_16x16x32_bf16(false, pf, false, vf,
                                                        (short)0, acc[dt], false, false);
    }
    __syncthreads();
  }

#pragma unroll
  for (int dt = 0; dt < 8; ++dt)
#pragma unroll
    for (int v = 0; v < 8; ++v) {
      int m = q0 + w * 16 + v + 8 * half;
      int col = h * 128 + dt * 16 + r;
      out[(size_t)m * 2048 + col] = acc[dt][v] / lrun[v];
    }
}

// ca[:, :2048] = attn ; ca[:, 2048:] = gelu(hbuf[:, 6144:14336])
__global__ __launch_bounds__(256) void gelu_concat(
    const float* __restrict__ attn, const float* __restrict__ hbuf,
    float* __restrict__ ca, int rows) {
  size_t idx = (size_t)blockIdx.x * 256 + threadIdx.x;
  size_t total = (size_t)rows * 10240;
  if (idx >= total) return;
  int m = (int)(idx / 10240), c = (int)(idx % 10240);
  float v;
  if (c < 2048) v = attn[(size_t)m * 2048 + c];
  else v = gelu_f(hbuf[(size_t)m * 14336 + 6144 + (c - 2048)]);
  ca[idx] = v;
}

// ---------------------------------------------------------------------------
// Host orchestration
// ---------------------------------------------------------------------------
static inline void launch_gemm(hipStream_t s, int act, const float* A, int lda,
                               const float* W, const float* bias, const float* res,
                               const float* gate, float* C, int ldc, int M, int N, int K) {
  dim3 g((N + 127) / 128, (M + 127) / 128), b(256);
  if (act == 1)      gemm_nt<1><<<g, b, 0, s>>>(A, lda, W, bias, res, gate, C, ldc, M, N, K);
  else if (act == 2) gemm_nt<2><<<g, b, 0, s>>>(A, lda, W, bias, res, gate, C, ldc, M, N, K);
  else               gemm_nt<0><<<g, b, 0, s>>>(A, lda, W, bias, res, gate, C, ldc, M, N, K);
}

extern "C" void kernel_launch(void* const* d_in, const int* in_sizes, int n_in,
                              void* d_out, int out_size, void* d_ws, size_t ws_size,
                              hipStream_t stream) {
  const int LT = 256, LI = 1024, LA = 1280, HD = 2048;
  const float* img = (const float*)d_in[0];   // 1024 x 64
  const float* txt = (const float*)d_in[1];   // 256 x 4096
  const float* apr = (const float*)d_in[2];   // 17 x 64
  const float* pe  = (const float*)d_in[3];   // 1280 x 64 x 2 x 2
  auto P = [&](int i) { return (const float*)d_in[i]; };
  // params: 4 img_in_w, 5 img_in_b, 6 txt_in_w, 7 txt_in_b, 8 appr_in_w, 9 appr_in_b,
  // 10+5l..14+5l: appr_l{l}_{in_w,in_b,out_w,out_b,norm}, 30 appr_out_w, 31 appr_out_b,
  // 32..41 d0_img_{qkv_w,qkv_b,qn,kn,proj_w,proj_b,mlp0_w,mlp0_b,mlp2_w,mlp2_b},
  // 42..51 d0_txt_ same, 52 s0_lin1_w, 53 s0_lin1_b, 54 s0_lin2_w, 55 s0_lin2_b,
  // 56 s0_qn, 57 s0_kn, 58 fl_w, 59 fl_b

  char* wsp = (char*)d_ws;
  auto alloc = [&](size_t bytes) -> void* {
    void* p = (void*)wsp;
    wsp += (bytes + 255) & ~(size_t)255;
    return p;
  };
  float*  mods = (float*)alloc((size_t)17 * HD * 4);
  float*  apx  = (float*)alloc((size_t)17 * HD * 4);
  float*  aph  = (float*)alloc((size_t)17 * HD * 4);
  float*  aph2 = (float*)alloc((size_t)17 * HD * 4);
  float*  img1 = (float*)alloc((size_t)LI * HD * 4);
  float*  txt1 = (float*)alloc((size_t)LT * HD * 4);
  float*  tmpA = (float*)alloc((size_t)LA * HD * 4);
  float*  bigA = (float*)alloc((size_t)LA * 14336 * 4);  // qkv_img/txt then hbuf
  float*  attn = (float*)alloc((size_t)LA * HD * 4);
  float*  mlp  = (float*)alloc((size_t)LA * 8192 * 4);
  float*  xbuf = (float*)alloc((size_t)LA * HD * 4);
  float*  ca   = (float*)alloc((size_t)LA * 10240 * 4);
  __bf16* qb   = (__bf16*)alloc((size_t)16 * LA * 128 * 2);
  __bf16* kb   = (__bf16*)alloc((size_t)16 * LA * 128 * 2);
  __bf16* vt   = (__bf16*)alloc((size_t)16 * LA * 128 * 2);
  float* qkv_img = bigA;
  float* qkv_txt = bigA + (size_t)LI * 6144;
  float* hbuf    = bigA;
  auto MV = [&](int i) { return mods + (size_t)i * HD; };

  // ---- input projections ----
  launch_gemm(stream, 0, img, 64,   P(4), P(5), nullptr, nullptr, img1, HD, LI, HD, 64);
  launch_gemm(stream, 0, txt, 4096, P(6), P(7), nullptr, nullptr, txt1, HD, LT, HD, 4096);

  // ---- approximator ----
  launch_gemm(stream, 0, apr, 64, P(8), P(9), nullptr, nullptr, apx, HD, 17, HD, 64);
  for (int l = 0; l < 4; ++l) {
    rms_row<<<17, 256, 0, stream>>>(apx, aph, P(14 + 5 * l), HD);
    launch_gemm(stream, 2, aph, HD, P(10 + 5 * l), P(11 + 5 * l), nullptr, nullptr, aph2, HD, 17, HD, HD);
    launch_gemm(stream, 0, aph2, HD, P(12 + 5 * l), P(13 + 5 * l), apx, nullptr, apx, HD, 17, HD, HD);
  }
  launch_gemm(stream, 0, apx, HD, P(30), P(31), nullptr, nullptr, mods, HD, 17, HD, HD);

  // ---- double-stream block: QKV ----
  ln_mod<<<LI, 256, 0, stream>>>(img1, tmpA, MV(4), MV(3), HD);
  launch_gemm(stream, 0, tmpA, HD, P(32), P(33), nullptr, nullptr, qkv_img, 6144, LI, 6144, HD);
  ln_mod<<<LT, 256, 0, stream>>>(txt1, tmpA, MV(10), MV(9), HD);
  launch_gemm(stream, 0, tmpA, HD, P(42), P(43), nullptr, nullptr, qkv_txt, 6144, LT, 6144, HD);

  // RMS + RoPE + bf16 pack (txt tokens at positions 0..255, img at 256..1279)
  qkv_post<<<dim3(LT, 16), 128, 0, stream>>>(qkv_txt, 6144, 0,   P(44), P(45), pe, qb, kb, vt, LA);
  qkv_post<<<dim3(LI, 16), 128, 0, stream>>>(qkv_img, 6144, 256, P(34), P(35), pe, qb, kb, vt, LA);
  flash_attn<<<dim3(LA / 64, 16), 128, 0, stream>>>(qb, kb, vt, attn, LA);

  // img branch: proj + MLP with gated residuals
  launch_gemm(stream, 0, attn + (size_t)LT * HD, HD, P(36), P(37), img1, MV(5), img1, HD, LI, HD, HD);
  ln_mod<<<LI, 256, 0, stream>>>(img1, tmpA, MV(7), MV(6), HD);
  launch_gemm(stream, 1, tmpA, HD, P(38), P(39), nullptr, nullptr, mlp, 8192, LI, 8192, HD);
  launch_gemm(stream, 0, mlp, 8192, P(40), P(41), img1, MV(8), xbuf + (size_t)LT * HD, HD, LI, HD, 8192);

  // txt branch
  launch_gemm(stream, 0, attn, HD, P(46), P(47), txt1, MV(11), txt1, HD, LT, HD, HD);
  ln_mod<<<LT, 256, 0, stream>>>(txt1, tmpA, MV(13), MV(12), HD);
  launch_gemm(stream, 1, tmpA, HD, P(48), P(49), nullptr, nullptr, mlp, 8192, LT, 8192, HD);
  launch_gemm(stream, 0, mlp, 8192, P(50), P(51), txt1, MV(14), xbuf, HD, LT, HD, 8192);

  // ---- single-stream block ----
  ln_mod<<<LA, 256, 0, stream>>>(xbuf, tmpA, MV(1), MV(0), HD);
  launch_gemm(stream, 0, tmpA, HD, P(52), P(53), nullptr, nullptr, hbuf, 14336, LA, 14336, HD);
  qkv_post<<<dim3(LA, 16), 128, 0, stream>>>(hbuf, 14336, 0, P(56), P(57), pe, qb, kb, vt, LA);
  flash_attn<<<dim3(LA / 64, 16), 128, 0, stream>>>(qb, kb, vt, attn, LA);
  {
    size_t total = (size_t)LA * 10240;
    gelu_concat<<<(unsigned)((total + 255) / 256), 256, 0, stream>>>(attn, hbuf, ca, LA);
  }
  launch_gemm(stream, 0, ca, 10240, P(54), P(55), xbuf, MV(2), xbuf, HD, LA, HD, 10240);

  // ---- final layer ----
  ln_mod<<<LI, 256, 0, stream>>>(xbuf + (size_t)LT * HD, tmpA, MV(16), MV(15), HD);
  launch_gemm(stream, 0, tmpA, HD, P(58), P(59), nullptr, nullptr, (float*)d_out, 64, LI, 64, HD);
}
